// TransformerVAE_27470610825961
// MI455X (gfx1250) — matrix operand order repaired and checked by reading, not compiled, not run on hardware
//
#include <hip/hip_runtime.h>

// ---------------------------------------------------------------------------
// TransformerVAE forward for gfx1250 (MI455X).  All matmuls run on
// v_wmma_f32_16x16x32_bf16 (bf16 A/B, fp32 accumulation).  Wave32 layouts per
// CDNA5 ISA:
//   A (16x32 bf16): lane<16 -> M=lane, K={0..7,16..23}; lane>=16 -> K={8..15,24..31}
//   B (32x16 bf16): lane<16 -> N=lane, K as for A; here A and B both use K=16*(lane>=16)+elem
//   C (16x16 f32) : vgpr r -> M = r + 8*(lane>=16), N = lane&15
// LDS tiles are stored so every fragment read is two 16B ds_load_b128s.
// bf16 staging copies use GLOBAL_LOAD_ASYNC_TO_LDS_B128 (ASYNCcnt) when the
// builtin is available; the GEMM double-buffers its LDS tiles so the next
// tile streams in while the current one feeds the WMMA pipe.
// ---------------------------------------------------------------------------

#define BB   8
#define LL   1024
#define DD   512
#define NHH  8
#define DKK  64
#define FFD  2048
#define TOK  (BB * LL)   // 8192 tokens

typedef __attribute__((ext_vector_type(16))) __bf16          v16bf;
typedef __attribute__((ext_vector_type(8)))  float           v8f;
typedef __attribute__((ext_vector_type(8)))  unsigned short  v8us;
typedef __attribute__((ext_vector_type(4)))  int             v4i;

#if __has_builtin(__builtin_amdgcn_global_load_async_to_lds_b128)
#define HAVE_ASYNC_LDS 1
#else
#define HAVE_ASYNC_LDS 0
#endif

__device__ __forceinline__ unsigned short f2bf(float f) {
    unsigned u = __builtin_bit_cast(unsigned, f);
    unsigned r = u + 0x7FFFu + ((u >> 16) & 1u);   // round-to-nearest-even
    return (unsigned short)(r >> 16);
}
__device__ __forceinline__ __bf16 bfb(unsigned short h) {
    return __builtin_bit_cast(__bf16, h);
}
__device__ __forceinline__ v8f wmma_bf16(v16bf a, v16bf b, v8f c) {
    return __builtin_amdgcn_wmma_f32_16x16x32_bf16(false, a, false, b,
                                                   (short)0, c, false, false);
}
// 16B/lane async copy global -> LDS (falls back to a plain vector copy).
__device__ __forceinline__ void async_cp16(const unsigned short* g, unsigned short* l) {
#if HAVE_ASYNC_LDS
    __builtin_amdgcn_global_load_async_to_lds_b128(
        (__attribute__((address_space(1))) v4i*)g,
        (__attribute__((address_space(3))) v4i*)l, 0, 0);
#else
    *(v8us*)l = *(const v8us*)g;
#endif
}
__device__ __forceinline__ void wait_async() {
#if HAVE_ASYNC_LDS
#if __has_builtin(__builtin_amdgcn_s_wait_asynccnt)
    __builtin_amdgcn_s_wait_asynccnt(0);
#else
    asm volatile("s_wait_asynccnt 0x0" ::: "memory");
#endif
#endif
}
// Load 16 contiguous bf16 (32B) from LDS as a fragment (2x ds_load_b128).
__device__ __forceinline__ v16bf ldfrag(const unsigned short* p) {
    union { v8us h[2]; v16bf f; } u;
    u.h[0] = *(const v8us*)p;
    u.h[1] = *(const v8us*)(p + 8);
    return u.f;
}
// Octet base position of the A-fragment permutation (oct = k>>3).
__device__ __forceinline__ int aoct(int oct) {
    return ((oct & 1) ? 16 : 0) + ((oct & 2) ? 8 : 0);
}
__device__ __forceinline__ float redsum32(float v) {
#pragma unroll
    for (int o = 16; o > 0; o >>= 1) v += __shfl_xor(v, o, 32);
    return v;
}
__device__ __forceinline__ float redmax16(float v) {
#pragma unroll
    for (int o = 8; o > 0; o >>= 1) v = fmaxf(v, __shfl_xor(v, o, 16));
    return v;
}
__device__ __forceinline__ float redsum16(float v) {
#pragma unroll
    for (int o = 8; o > 0; o >>= 1) v += __shfl_xor(v, o, 16);
    return v;
}
__device__ __forceinline__ void wavebar() {
#if __has_builtin(__builtin_amdgcn_wave_barrier)
    __builtin_amdgcn_wave_barrier();
#else
    __syncthreads();
#endif
}

// ---------------------------------------------------------------------------
// Relative position bias table: bias[h][i][j]
// ---------------------------------------------------------------------------
__global__ __launch_bounds__(256) void bias_kernel(const float* __restrict__ rpb,
                                                   float* __restrict__ bias) {
    int idx = blockIdx.x * 256 + threadIdx.x;          // 0 .. L*L-1
    int i = idx >> 10, j = idx & 1023;
    int rp  = j - i;
    int ret = (rp < 0) ? 32 : 0;
    int arp = (rp < 0) ? -rp : rp;
    int bucket;
    if (arp < 16) {
        bucket = ret + arp;
    } else {
        int vl = 16 + (int)(logf((float)arp / 16.0f + 1e-6f) * (16.0f / logf(16.0f)));
        vl = min(vl, 31);
        bucket = ret + vl;
    }
#pragma unroll
    for (int h = 0; h < NHH; h++)
        bias[(size_t)h * LL * LL + idx] = rpb[bucket * NHH + h];
}

// ---------------------------------------------------------------------------
// Input embed (IN=4) + ReLU + LayerNorm.  One wave per token row.
// ---------------------------------------------------------------------------
__global__ __launch_bounds__(256) void embed_kernel(const float* __restrict__ x,
                                                    const float* __restrict__ w,
                                                    const float* __restrict__ b,
                                                    const float* __restrict__ g,
                                                    const float* __restrict__ bn,
                                                    float* __restrict__ out) {
    int t    = blockIdx.x * 8 + (threadIdx.x >> 5);
    int lane = threadIdx.x & 31;
    float x0 = x[t * 4 + 0], x1 = x[t * 4 + 1], x2 = x[t * 4 + 2], x3 = x[t * 4 + 3];
    float vals[16];
    float s = 0.f;
#pragma unroll
    for (int j = 0; j < 16; j++) {
        int d = lane + 32 * j;
        float v = x0 * w[d] + x1 * w[DD + d] + x2 * w[2 * DD + d] + x3 * w[3 * DD + d] + b[d];
        v = fmaxf(v, 0.f);
        vals[j] = v; s += v;
    }
    float mean = redsum32(s) * (1.0f / DD);
    float var = 0.f;
#pragma unroll
    for (int j = 0; j < 16; j++) { float d = vals[j] - mean; var += d * d; }
    var = redsum32(var) * (1.0f / DD);
    float inv = rsqrtf(var + 1e-5f);
    float* yr = out + (size_t)t * DD;
#pragma unroll
    for (int j = 0; j < 16; j++) {
        int d = lane + 32 * j;
        yr[d] = (vals[j] - mean) * inv * g[d] + bn[d];
    }
}

// ---------------------------------------------------------------------------
// LayerNorm: one wave per token row.
// ---------------------------------------------------------------------------
__global__ __launch_bounds__(256) void ln_kernel(const float* __restrict__ x,
                                                 const float* __restrict__ g,
                                                 const float* __restrict__ b,
                                                 float* __restrict__ y) {
    int t    = blockIdx.x * 8 + (threadIdx.x >> 5);
    int lane = threadIdx.x & 31;
    const float* xr = x + (size_t)t * DD;
    float vals[16];
    float s = 0.f;
#pragma unroll
    for (int j = 0; j < 16; j++) { vals[j] = xr[lane + 32 * j]; s += vals[j]; }
    float mean = redsum32(s) * (1.0f / DD);
    float var = 0.f;
#pragma unroll
    for (int j = 0; j < 16; j++) { float d = vals[j] - mean; var += d * d; }
    var = redsum32(var) * (1.0f / DD);
    float inv = rsqrtf(var + 1e-5f);
    float* yr = y + (size_t)t * DD;
#pragma unroll
    for (int j = 0; j < 16; j++) {
        int d = lane + 32 * j;
        yr[d] = (vals[j] - mean) * inv * g[d] + b[d];
    }
}

// ---------------------------------------------------------------------------
// Generic WMMA GEMM: out = epilogue(alpha * A[M,K] @ W[K,N] + bias) + res
// 256 threads = 8 waves (4x2), block tile 64x64, wave tile 16x32, K step 32.
// Double-buffered LDS: stage tile k+1 (async A when bf16) while computing k.
// Template: ABF = A is bf16, OBF = out is bf16, ACT = 1 -> exact GELU.
// ---------------------------------------------------------------------------
template <int ABF, int OBF, int ACT>
__global__ __launch_bounds__(256) void gemm_kernel(const void* __restrict__ Ap,
                                                   const float* __restrict__ W,
                                                   const float* __restrict__ bias,
                                                   const float* __restrict__ res,
                                                   void* __restrict__ outp,
                                                   int M, int N, int K,
                                                   float alpha) {
    __shared__ unsigned short As[2][64][40];   // [buf][row][k]   stride 80B
    __shared__ unsigned short Bs[2][64][40];   // [buf][n][k]        stride 80B
    int tid  = threadIdx.x, wvid = tid >> 5, lane = tid & 31;
    int hi   = lane >> 4,  l15  = lane & 15;
    int wm   = wvid >> 1,  wn   = wvid & 1;
    int row0 = blockIdx.y * 64, col0 = blockIdx.x * 64;
    // staging coordinates (one octet per thread for A; packed pairs for W)
    int ar   = tid >> 2, aoctn = tid & 3, ac = aoctn * 8, apos = ac;
    const float*          Af = (const float*)Ap;
    const unsigned short* Ah = (const unsigned short*)Ap;

    auto stage = [&](int buf, int kk) {
        if constexpr (ABF) {
            async_cp16(&Ah[(size_t)(row0 + ar) * K + kk + ac], &As[buf][ar][apos]);
        } else {
            const float* src = &Af[(size_t)(row0 + ar) * K + kk + ac];
            v8us pk;
#pragma unroll
            for (int i = 0; i < 8; i++) pk[i] = f2bf(src[i]);
            *(v8us*)&As[buf][ar][apos] = pk;
        }
#pragma unroll
        for (int t = 0; t < 4; t++) {                  // W 32x64 transposed
            int idx = t * 256 + tid;                   // 0..1023
            int n = idx & 63, k2 = (idx >> 6) << 1;    // k2 = 0,2,..,30
            float w0 = W[(size_t)(kk + k2) * N + col0 + n];
            float w1 = W[(size_t)(kk + k2 + 1) * N + col0 + n];
            unsigned pkw = (unsigned)f2bf(w0) | ((unsigned)f2bf(w1) << 16);
            *(unsigned*)&Bs[buf][n][k2] = pkw;
        }
    };

    v8f acc[2] = {};
    stage(0, 0);
    wait_async();
    __syncthreads();
    int pb = 0;
    for (int k0 = 0; k0 < K; k0 += 32) {
        int nb = pb ^ 1;
        if (k0 + 32 < K) stage(nb, k0 + 32);           // overlap with compute
        v16bf a  = ldfrag(&As[pb][wm * 16 + l15][hi * 16]);
        v16bf b0 = ldfrag(&Bs[pb][wn * 32 + l15][hi * 16]);
        v16bf b1 = ldfrag(&Bs[pb][wn * 32 + 16 + l15][hi * 16]);
        acc[0] = wmma_bf16(a, b0, acc[0]);
        acc[1] = wmma_bf16(a, b1, acc[1]);
        wait_async();
        __syncthreads();
        pb = nb;
    }
#pragma unroll
    for (int t = 0; t < 2; t++) {
        int col = col0 + wn * 32 + t * 16 + l15;
#pragma unroll
        for (int r = 0; r < 8; r++) {
            int row = row0 + wm * 16 + hi * 8 + r;
            float vv = acc[t][r] * alpha;
            if (bias) vv += bias[col];
            if constexpr (ACT == 1)
                vv = 0.5f * vv * (1.0f + erff(vv * 0.70710678118654752f));
            if (res) vv += res[(size_t)row * N + col];
            size_t oi = (size_t)row * N + col;
            if constexpr (OBF) ((unsigned short*)outp)[oi] = f2bf(vv);
            else               ((float*)outp)[oi] = vv;
        }
    }
}

// ---------------------------------------------------------------------------
// Flash-style attention.  Block = 8 waves; all waves share one (b,h), each
// wave owns 16 queries.  q/k/v/ctx are bf16 (q pre-scaled by 1/sqrt(DK)).
// ---------------------------------------------------------------------------
__global__ __launch_bounds__(256) void attn_kernel(const unsigned short* __restrict__ q,
                                                   const unsigned short* __restrict__ k,
                                                   const unsigned short* __restrict__ v,
                                                   const float* __restrict__ bias,
                                                   unsigned short* __restrict__ ctx) {
    __shared__ unsigned short Kt[32][72];      // [key][d]   stride 144B
    __shared__ unsigned short Vt[64][40];      // [d][key]   stride 80B
    __shared__ unsigned short Pb[8][16][40];   // per-wave P, [row][key]
    int tid  = threadIdx.x, wv = tid >> 5, lane = tid & 31;
    int hi   = lane >> 4,  l15 = lane & 15;
    int b    = blockIdx.x >> 3, h = blockIdx.x & 7;
    int qbase = blockIdx.y * 128 + wv * 16;
    const unsigned short* qp = q + (size_t)b * LL * DD + h * DKK;
    const unsigned short* kp = k + (size_t)b * LL * DD + h * DKK;
    const unsigned short* vp = v + (size_t)b * LL * DD + h * DKK;
    unsigned short*       cp = ctx + (size_t)b * LL * DD + h * DKK;
    const float*          bp = bias + (size_t)h * LL * LL;
    int skey = tid >> 3, sd = (tid & 7) * 8;   // staging coords (32 keys x 8 octets)

    v16bf aq0, aq1;                            // Q tile 16x64 as two A-frags
    {
        const unsigned short* qr = qp + (size_t)(qbase + l15) * DD;
#pragma unroll
        for (int i = 0; i < 16; i++) {
            int kk = hi * 16 + i;
            aq0[i] = bfb(qr[kk]);
            aq1[i] = bfb(qr[32 + kk]);
        }
    }
    float m[8], lsum[8];
    v8f o[4] = {};
#pragma unroll
    for (int r = 0; r < 8; r++) { m[r] = -1e30f; lsum[r] = 0.f; }

    for (int kb = 0; kb < LL; kb += 32) {
        // ---- stage K (async b128 copy) and V^T (scatter on store side) ----
        async_cp16(&kp[(size_t)(kb + skey) * DD + sd], &Kt[skey][sd]);
        {
            v8us vv = *(const v8us*)&vp[(size_t)(kb + skey) * DD + sd];
#pragma unroll
            for (int i = 0; i < 8; i++) Vt[sd + i][skey] = vv[i];
        }
        wait_async();
        __syncthreads();
        // ---- S = Q K^T (two 16-key subtiles) ----
        v8f s0 = {}, s1 = {};
        {
            v16bf b00 = ldfrag(&Kt[l15][hi * 16]);
            v16bf b01 = ldfrag(&Kt[l15][32 + hi * 16]);
            s0 = wmma_bf16(aq0, b00, s0);
            s0 = wmma_bf16(aq1, b01, s0);
        }
        {
            v16bf b10 = ldfrag(&Kt[16 + l15][hi * 16]);
            v16bf b11 = ldfrag(&Kt[16 + l15][32 + hi * 16]);
            s1 = wmma_bf16(aq0, b10, s1);
            s1 = wmma_bf16(aq1, b11, s1);
        }
#pragma unroll
        for (int r = 0; r < 8; r++) {          // + relative position bias
            int qpos = qbase + hi * 8 + r;
            s0[r] += bp[(size_t)qpos * LL + kb + l15];
            s1[r] += bp[(size_t)qpos * LL + kb + 16 + l15];
        }
        int pp0 = l15;                                    // key l15
        int pp1 = l15 + 16;                               // key 16+l15
#pragma unroll
        for (int r = 0; r < 8; r++) {          // online softmax per row
            float mx = redmax16(fmaxf(s0[r], s1[r]));
            float mn = fmaxf(m[r], mx);
            float f  = expf(m[r] - mn);
            float p0 = expf(s0[r] - mn);
            float p1 = expf(s1[r] - mn);
            lsum[r] = lsum[r] * f + redsum16(p0 + p1);
            o[0][r] *= f; o[1][r] *= f; o[2][r] *= f; o[3][r] *= f;
            m[r] = mn;
            Pb[wv][hi * 8 + r][pp0] = f2bf(p0);
            Pb[wv][hi * 8 + r][pp1] = f2bf(p1);
        }
        wavebar();
        v16bf ap = ldfrag(&Pb[wv][l15][hi * 16]);
        wavebar();
#pragma unroll
        for (int t = 0; t < 4; t++) {          // O += P @ V
            v16bf bv = ldfrag(&Vt[t * 16 + l15][hi * 16]);
            o[t] = wmma_bf16(ap, bv, o[t]);
        }
        __syncthreads();                       // all reads done before restage
    }
#pragma unroll
    for (int t = 0; t < 4; t++) {
#pragma unroll
        for (int r = 0; r < 8; r++) {
            int qpos = qbase + hi * 8 + r;
            cp[(size_t)qpos * DD + t * 16 + l15] = f2bf(o[t][r] / lsum[r]);
        }
    }
}

// ---------------------------------------------------------------------------
// z = mu + eps * exp(0.5*logvar)
// ---------------------------------------------------------------------------
__global__ __launch_bounds__(256) void reparam_kernel(const float* __restrict__ mu,
                                                      const float* __restrict__ lv,
                                                      const float* __restrict__ eps,
                                                      float* __restrict__ z) {
    size_t i = (size_t)blockIdx.x * 256 + threadIdx.x;
    z[i] = mu[i] + eps[i] * expf(0.5f * lv[i]);
}

// ---------------------------------------------------------------------------
// px[b,l,m,{0,1}] = {-s, s},  s = q[b,l,:]·k[b,m,:] / sqrt(D).  q/k bf16.
// ---------------------------------------------------------------------------
__global__ __launch_bounds__(256) void score_kernel(const unsigned short* __restrict__ q,
                                                    const unsigned short* __restrict__ k,
                                                    float* __restrict__ px) {
    __shared__ unsigned short Qs[64][40];      // [row][d]
    __shared__ unsigned short Ks[64][40];      // [m][d]
    int tid  = threadIdx.x, wvid = tid >> 5, lane = tid & 31;
    int hi   = lane >> 4,  l15  = lane & 15;
    int wm   = wvid >> 1,  wn   = wvid & 1;
    int b    = blockIdx.z;
    int row0 = blockIdx.y * 64;
    int col0 = blockIdx.x * 64;
    int sr   = tid >> 2, soct = tid & 3, sc = soct * 8, spos = sc;
    const unsigned short* qp = q + (size_t)b * LL * DD;
    const unsigned short* kp = k + (size_t)b * LL * DD;
    v8f c[2] = {};
    for (int d0 = 0; d0 < DD; d0 += 32) {
        async_cp16(&qp[(size_t)(row0 + sr) * DD + d0 + sc], &Qs[sr][spos]);
        async_cp16(&kp[(size_t)(col0 + sr) * DD + d0 + sc], &Ks[sr][sc]);
        wait_async();
        __syncthreads();
        v16bf a  = ldfrag(&Qs[wm * 16 + l15][hi * 16]);
        v16bf b0 = ldfrag(&Ks[wn * 32 + l15][hi * 16]);
        v16bf b1 = ldfrag(&Ks[wn * 32 + 16 + l15][hi * 16]);
        c[0] = wmma_bf16(a, b0, c[0]);
        c[1] = wmma_bf16(a, b1, c[1]);
        __syncthreads();
    }
    const float sc2 = 0.04419417382415922f;    // 1/sqrt(512)
#pragma unroll
    for (int t = 0; t < 2; t++) {
#pragma unroll
        for (int r = 0; r < 8; r++) {
            int row = row0 + wm * 16 + hi * 8 + r;
            int col = col0 + wn * 32 + t * 16 + l15;
            float sv = c[t][r] * sc2;
            size_t oidx = (((size_t)b * LL + row) * LL + col) * 2;
            float2 pr; pr.x = -sv; pr.y = sv;
            *(float2*)&px[oidx] = pr;          // global_store_b64
        }
    }
}

// ---------------------------------------------------------------------------
// Host-side orchestration
// ---------------------------------------------------------------------------
template <int ABF, int OBF, int ACT>
static inline void run_gemm(const void* A, const float* W, const float* bias,
                            const float* res, void* out, int M, int N, int K,
                            float alpha, hipStream_t s) {
    dim3 grid(N / 64, M / 64);
    gemm_kernel<ABF, OBF, ACT><<<grid, 256, 0, s>>>(A, W, bias, res, out, M, N, K, alpha);
}

static void run_layer(void* const* d_in, int base, float** pxa, float** pxb,
                      float* xn, unsigned short* qb, unsigned short* kb,
                      unsigned short* vb, unsigned short* cb, unsigned short* ffb,
                      const float* biasb, hipStream_t s) {
    const float* n1_g = (const float*)d_in[base + 0];
    const float* n1_b = (const float*)d_in[base + 1];
    const float* wq   = (const float*)d_in[base + 2];
    const float* wk   = (const float*)d_in[base + 3];
    const float* wv   = (const float*)d_in[base + 4];
    const float* wo   = (const float*)d_in[base + 5];
    const float* n2_g = (const float*)d_in[base + 6];
    const float* n2_b = (const float*)d_in[base + 7];
    const float* w1   = (const float*)d_in[base + 8];
    const float* b1   = (const float*)d_in[base + 9];
    const float* w2   = (const float*)d_in[base + 10];
    const float* b2   = (const float*)d_in[base + 11];
    float* xa = *pxa;
    float* xb = *pxb;

    ln_kernel<<<TOK / 8, 256, 0, s>>>(xa, n1_g, n1_b, xn);
    run_gemm<0, 1, 0>(xn, wq, nullptr, nullptr, qb, TOK, DD, DD, 0.125f, s);
    run_gemm<0, 1, 0>(xn, wk, nullptr, nullptr, kb, TOK, DD, DD, 1.0f, s);
    run_gemm<0, 1, 0>(xn, wv, nullptr, nullptr, vb, TOK, DD, DD, 1.0f, s);
    attn_kernel<<<dim3(BB * NHH, LL / 128), 256, 0, s>>>(qb, kb, vb, biasb, cb);
    run_gemm<1, 0, 0>(cb, wo, nullptr, xa, xb, TOK, DD, DD, 1.0f, s);  // x += ctx@wo
    { float* t = xa; xa = xb; xb = t; }
    ln_kernel<<<TOK / 8, 256, 0, s>>>(xa, n2_g, n2_b, xn);
    run_gemm<0, 1, 1>(xn, w1, b1, nullptr, ffb, TOK, FFD, DD, 1.0f, s);   // GELU
    run_gemm<1, 0, 0>(ffb, w2, b2, xa, xb, TOK, DD, FFD, 1.0f, s);        // x += ff@w2+b2
    { float* t = xa; xa = xb; xb = t; }
    *pxa = xa;
    *pxb = xb;
}

extern "C" void kernel_launch(void* const* d_in, const int* in_sizes, int n_in,
                              void* d_out, int out_size, void* d_ws, size_t ws_size,
                              hipStream_t stream) {
    (void)in_sizes; (void)n_in; (void)out_size; (void)ws_size;
    // Input order = setup_inputs() insertion order, params flattened depth-first:
    // 0:x 1:eps 2:in_w 3:in_b 4:in_g 5:in_bn 6:rpb
    // 7..54: enc (4 layers x 12)  55:mu_w 56:mu_b 57:lv_w 58:lv_b
    // 59..106: dec (4 layers x 12) 107:q_w 108:q_b 109:k_w 110:k_b
    const float* x     = (const float*)d_in[0];
    const float* eps   = (const float*)d_in[1];
    const float* in_w  = (const float*)d_in[2];
    const float* in_b  = (const float*)d_in[3];
    const float* in_g  = (const float*)d_in[4];
    const float* in_bn = (const float*)d_in[5];
    const float* rpb   = (const float*)d_in[6];
    const int ENC0 = 7, DEC0 = 59;
    const float* mu_w = (const float*)d_in[55];
    const float* mu_b = (const float*)d_in[56];
    const float* lv_w = (const float*)d_in[57];
    const float* lv_b = (const float*)d_in[58];
    const float* q_w  = (const float*)d_in[107];
    const float* q_b  = (const float*)d_in[108];
    const float* k_w  = (const float*)d_in[109];
    const float* k_b  = (const float*)d_in[110];

    float* out = (float*)d_out;
    float* px  = out;                                   // [B,L,L,2]
    float* mu  = out + (size_t)BB * LL * LL * 2;        // [B,L,D]
    float* lv  = mu  + (size_t)TOK * DD;                // [B,L,D]

    char* wsb = (char*)d_ws;
    size_t off = 0;
    auto alloc = [&](size_t bytes) -> void* {
        void* p = wsb + off;
        off += (bytes + 255) & ~(size_t)255;
        return p;
    };
    float*          biasb = (float*)alloc((size_t)NHH * LL * LL * 4);
    float*          xa    = (float*)alloc((size_t)TOK * DD * 4);
    float*          xb    = (float*)alloc((size_t)TOK * DD * 4);
    float*          xn    = (float*)alloc((size_t)TOK * DD * 4);
    unsigned short* qb    = (unsigned short*)alloc((size_t)TOK * DD * 2);
    unsigned short* kb    = (unsigned short*)alloc((size_t)TOK * DD * 2);
    unsigned short* vb    = (unsigned short*)alloc((size_t)TOK * DD * 2);
    unsigned short* cb    = (unsigned short*)alloc((size_t)TOK * DD * 2);
    unsigned short* ffb   = (unsigned short*)alloc((size_t)TOK * FFD * 2);

    bias_kernel<<<(LL * LL) / 256, 256, 0, stream>>>(rpb, biasb);
    embed_kernel<<<TOK / 8, 256, 0, stream>>>(x, in_w, in_b, in_g, in_bn, xa);

    for (int l = 0; l < 4; l++)
        run_layer(d_in, ENC0 + 12 * l, &xa, &xb, xn, qb, kb, vb, cb, ffb, biasb, stream);

    run_gemm<0, 0, 0>(xa, mu_w, mu_b, nullptr, mu, TOK, DD, DD, 1.0f, stream);
    run_gemm<0, 0, 0>(xa, lv_w, lv_b, nullptr, lv, TOK, DD, DD, 1.0f, stream);
    reparam_kernel<<<((size_t)TOK * DD) / 256, 256, 0, stream>>>(mu, lv, eps, xb);
    { float* t = xa; xa = xb; xb = t; }                 // xa = z

    for (int l = 0; l < 4; l++)
        run_layer(d_in, DEC0 + 12 * l, &xa, &xb, xn, qb, kb, vb, cb, ffb, biasb, stream);

    run_gemm<0, 1, 0>(xa, q_w, q_b, nullptr, qb, TOK, DD, DD, 1.0f, stream);
    run_gemm<0, 1, 0>(xa, k_w, k_b, nullptr, kb, TOK, DD, DD, 1.0f, stream);
    score_kernel<<<dim3(LL / 64, LL / 64, BB), 256, 0, stream>>>(qb, kb, px);
}
